// GCMC_35519379538608
// MI455X (gfx1250) — compile-verified, matrix operand order checked by reading, not run on hardware
//
#include <hip/hip_runtime.h>
#include <cstdint>
#include <cstddef>

typedef __attribute__((ext_vector_type(16))) __bf16 v16bf;
typedef __attribute__((ext_vector_type(8)))  __bf16 v8bf;
typedef __attribute__((ext_vector_type(8)))  float  v8f;

__device__ __forceinline__ __bf16 f2bf(float f) {
  union { float f; unsigned u; } in; in.f = f;
  unsigned r = (in.u + 0x7FFFu + ((in.u >> 16) & 1u)) >> 16;   // RNE
  union { unsigned short s; __bf16 b; } out; out.s = (unsigned short)r;
  return out.b;
}

// ---------------------------------------------------------------------------
// Elementwise converters
// ---------------------------------------------------------------------------
__global__ __launch_bounds__(256) void cvt_f2bf(const float* __restrict__ src,
                                                __bf16* __restrict__ dst, int n) {
  int g = blockIdx.x * 256 + threadIdx.x;
  if (g < n) dst[g] = f2bf(src[g]);
}

// src [R,C] row-major -> dst [C,R] row-major (i.e., dst[c,r] = src[r,c])
__global__ __launch_bounds__(256) void transpose_f2bf(const float* __restrict__ src,
                                                      __bf16* __restrict__ dst,
                                                      int R, int C) {
  int g = blockIdx.x * 256 + threadIdx.x;
  if (g < R * C) {
    int r = g / C, c = g % C;
    dst[(size_t)c * R + r] = f2bf(src[g]);
  }
}

// cat[m, 0:256)   = bf16(relu(acc[m,:]))
// cat[m, 256:512) = bf16(feat[m,:])
__global__ __launch_bounds__(256) void pack_cat(const float* __restrict__ acc,
                                                const float* __restrict__ feat,
                                                __bf16* __restrict__ cat, int M) {
  int g = blockIdx.x * 256 + threadIdx.x;
  if (g >= M * 256) return;
  int m = g >> 8, c = g & 255;
  float h = acc[g];
  cat[(size_t)m * 512 + c]       = f2bf(h > 0.0f ? h : 0.0f);
  cat[(size_t)m * 512 + 256 + c] = f2bf(feat[g]);
}

// ---------------------------------------------------------------------------
// COO SPMM scatter: acc[rows[e],:] += vals[e] * x[cols[e],:]   (D = 256)
// 64 threads per edge, 4 floats each, f32 global atomics.
// ---------------------------------------------------------------------------
__global__ __launch_bounds__(256) void spmm_scatter(const int* __restrict__ rows,
                                                    const int* __restrict__ cols,
                                                    const float* __restrict__ vals,
                                                    const float* __restrict__ x,
                                                    float* __restrict__ acc, int nEdges) {
  int g = blockIdx.x * 256 + threadIdx.x;
  int e = g >> 6;
  if (e >= nEdges) return;
  int c4 = (g & 63) * 4;
  float v = vals[e];
  const float4 xv = *(const float4*)&x[(size_t)cols[e] * 256 + c4];
  float* p = &acc[(size_t)rows[e] * 256 + c4];
  __hip_atomic_fetch_add(p + 0, v * xv.x, __ATOMIC_RELAXED, __HIP_MEMORY_SCOPE_AGENT);
  __hip_atomic_fetch_add(p + 1, v * xv.y, __ATOMIC_RELAXED, __HIP_MEMORY_SCOPE_AGENT);
  __hip_atomic_fetch_add(p + 2, v * xv.z, __ATOMIC_RELAXED, __HIP_MEMORY_SCOPE_AGENT);
  __hip_atomic_fetch_add(p + 3, v * xv.w, __ATOMIC_RELAXED, __HIP_MEMORY_SCOPE_AGENT);
}

// ---------------------------------------------------------------------------
// BatchNorm: per-column sum / sumsq over h[M,128]; one block per column.
// ---------------------------------------------------------------------------
__global__ __launch_bounds__(256) void bn_stats(const float* __restrict__ h, int M,
                                                float* __restrict__ stats /*[256]*/) {
  int col = blockIdx.x;                 // 0..127
  float s = 0.0f, s2 = 0.0f;
  for (int r = threadIdx.x; r < M; r += 256) {
    float x = h[(size_t)r * 128 + col];
    s += x; s2 += x * x;
  }
  __shared__ float sh[256], sh2[256];
  sh[threadIdx.x] = s; sh2[threadIdx.x] = s2;
  __syncthreads();
  for (int o = 128; o > 0; o >>= 1) {
    if (threadIdx.x < o) { sh[threadIdx.x] += sh[threadIdx.x + o];
                           sh2[threadIdx.x] += sh2[threadIdx.x + o]; }
    __syncthreads();
  }
  if (threadIdx.x == 0) { stats[col] = sh[0]; stats[128 + col] = sh2[0]; }
}

__global__ __launch_bounds__(256) void bn_apply(const float* __restrict__ h,
                                                const float* __restrict__ stats,
                                                const float* __restrict__ gam,
                                                const float* __restrict__ beta,
                                                __bf16* __restrict__ e, int M) {
  int g = blockIdx.x * 256 + threadIdx.x;
  if (g >= M * 128) return;
  int c = g & 127;
  float inv = 1.0f / (float)M;
  float mu  = stats[c] * inv;
  float var = stats[128 + c] * inv - mu * mu;
  float y = (h[g] - mu) * rsqrtf(var + 1e-5f) * gam[c] + beta[c];
  e[g] = f2bf(y > 0.0f ? y : 0.0f);
}

// ---------------------------------------------------------------------------
// bf16 NT GEMM with f32 accumulate via v_wmma_f32_16x16x32_bf16.
// C[m,n] = sum_k A[m,k] * B[n,k]  (+ bias[n])
// Block 256 thr (8 waves), block tile 128x64, wave tile 32x32 (4 WMMA / K-slice).
// outF (f32) and/or outB (bf16) may be null. M%128==0, N%64==0, K%32==0.
// ---------------------------------------------------------------------------
__device__ __forceinline__ void store_tile(const v8f c, int m0, int n,
                                           float* __restrict__ outF,
                                           __bf16* __restrict__ outB,
                                           const float* __restrict__ bias, int N) {
  float bv = bias ? bias[n] : 0.0f;
#pragma unroll
  for (int v = 0; v < 8; ++v) {
    float val = c[v] + bv;
    size_t off = (size_t)(m0 + v) * (size_t)N + n;
    if (outF) outF[off] = val;
    if (outB) outB[off] = f2bf(val);
  }
}

__global__ __launch_bounds__(256) void gemm_nt_bf16(const __bf16* __restrict__ A,
                                                    const __bf16* __restrict__ B,
                                                    float* __restrict__ outF,
                                                    __bf16* __restrict__ outB,
                                                    const float* __restrict__ bias,
                                                    int M, int N, int K) {
  __shared__ __bf16 lA[128 * 32];
  __shared__ __bf16 lB[64 * 32];
  const int tid   = threadIdx.x;
  const int wave  = tid >> 5;
  const int lane  = tid & 31;
  const int tileM = blockIdx.y * 128;
  const int tileN = blockIdx.x * 64;
  const int wM    = (wave >> 1) * 32;
  const int wN    = (wave & 1) * 32;
  const int row16 = lane & 15;
  const int kg    = (lane >> 4) * 8;     // K-half select per ISA 16-bit A/B layout
  const int lr    = tid >> 2;            // LDS fill: row
  const int lc    = (tid & 3) * 8;       // LDS fill: 8-elem chunk

  v8f c00 = {}, c01 = {}, c10 = {}, c11 = {};

  for (int k0 = 0; k0 < K; k0 += 32) {
    *(v8bf*)&lA[(size_t)lr * 32 + lc] =
        *(const v8bf*)&A[(size_t)(tileM + lr) * K + k0 + lc];
    *(v8bf*)&lA[(size_t)(lr + 64) * 32 + lc] =
        *(const v8bf*)&A[(size_t)(tileM + lr + 64) * K + k0 + lc];
    *(v8bf*)&lB[(size_t)lr * 32 + lc] =
        *(const v8bf*)&B[(size_t)(tileN + lr) * K + k0 + lc];
    if (k0 + 32 < K)
      __builtin_prefetch(&A[(size_t)(tileM + lr) * K + k0 + 32 + lc], 0, 1);
    __syncthreads();

    v16bf a0, a1, b0, b1;
    {
      const __bf16* p; v8bf t;
      p = &lA[(size_t)(wM + row16) * 32 + kg];
      t = *(const v8bf*)p;
#pragma unroll
      for (int i = 0; i < 8; ++i) a0[i] = t[i];
      t = *(const v8bf*)(p + 16);
#pragma unroll
      for (int i = 0; i < 8; ++i) a0[i + 8] = t[i];
      p = &lA[(size_t)(wM + 16 + row16) * 32 + kg];
      t = *(const v8bf*)p;
#pragma unroll
      for (int i = 0; i < 8; ++i) a1[i] = t[i];
      t = *(const v8bf*)(p + 16);
#pragma unroll
      for (int i = 0; i < 8; ++i) a1[i + 8] = t[i];
      p = &lB[(size_t)(wN + row16) * 32 + kg];
      t = *(const v8bf*)p;
#pragma unroll
      for (int i = 0; i < 8; ++i) b0[i] = t[i];
      t = *(const v8bf*)(p + 16);
#pragma unroll
      for (int i = 0; i < 8; ++i) b0[i + 8] = t[i];
      p = &lB[(size_t)(wN + 16 + row16) * 32 + kg];
      t = *(const v8bf*)p;
#pragma unroll
      for (int i = 0; i < 8; ++i) b1[i] = t[i];
      t = *(const v8bf*)(p + 16);
#pragma unroll
      for (int i = 0; i < 8; ++i) b1[i + 8] = t[i];
    }
    c00 = __builtin_amdgcn_wmma_f32_16x16x32_bf16(false, a0, false, b0, (short)0, c00, false, false);
    c01 = __builtin_amdgcn_wmma_f32_16x16x32_bf16(false, a0, false, b1, (short)0, c01, false, false);
    c10 = __builtin_amdgcn_wmma_f32_16x16x32_bf16(false, a1, false, b0, (short)0, c10, false, false);
    c11 = __builtin_amdgcn_wmma_f32_16x16x32_bf16(false, a1, false, b1, (short)0, c11, false, false);
    __syncthreads();
  }

  // C/D layout: VGPR v, lanes 0-15 -> M=v, N=lane; lanes 16-31 -> M=8+v, N=lane-16
  const int subM = (lane >> 4) * 8;
  const int col  = lane & 15;
  const int mB = tileM + wM + subM;
  const int nB = tileN + wN + col;
  store_tile(c00, mB,      nB,      outF, outB, bias, N);
  store_tile(c01, mB,      nB + 16, outF, outB, bias, N);
  store_tile(c10, mB + 16, nB,      outF, outB, bias, N);
  store_tile(c11, mB + 16, nB + 16, outF, outB, bias, N);
}

// ---------------------------------------------------------------------------
// Host orchestration
// ---------------------------------------------------------------------------
extern "C" void kernel_launch(void* const* d_in, const int* in_sizes, int n_in,
                              void* d_out, int out_size, void* d_ws, size_t ws_size,
                              hipStream_t stream) {
  (void)n_in; (void)out_size; (void)ws_size;
  const int D = 256, OUT = 128, K2 = 512;

  const float* fq    = (const float*)d_in[0];
  const float* fi    = (const float*)d_in[1];
  const float* ft    = (const float*)d_in[2];
  const int*   qi_src = (const int*)d_in[3];
  const int*   qi_dst = (const int*)d_in[4];
  const float* qi_val = (const float*)d_in[5];
  const int*   it_src = (const int*)d_in[6];
  const int*   it_dst = (const int*)d_in[7];
  const float* it_val = (const float*)d_in[8];
  const float* W     = (const float*)d_in[9];
  const float* Wq    = (const float*)d_in[10];
  const float* bq    = (const float*)d_in[11];
  const float* gq    = (const float*)d_in[12];
  const float* betaq = (const float*)d_in[13];
  const float* Wi    = (const float*)d_in[14];
  const float* bi    = (const float*)d_in[15];
  const float* gi    = (const float*)d_in[16];
  const float* betai = (const float*)d_in[17];
  const float* Wt    = (const float*)d_in[18];
  const float* bt    = (const float*)d_in[19];
  const float* gt    = (const float*)d_in[20];
  const float* betat = (const float*)d_in[21];
  const float* Q     = (const float*)d_in[22];

  const int NQ  = in_sizes[0] / D;
  const int NI  = in_sizes[1] / D;
  const int NT  = in_sizes[2] / D;
  const int EQI = in_sizes[3];
  const int EIT = in_sizes[6];

  char* ws = (char*)d_ws;
  size_t off = 0;
  auto alloc = [&](size_t bytes) -> void* {
    off = (off + 255) & ~(size_t)255;
    void* p = ws + off;
    off += bytes;
    return p;
  };

  __bf16* fq_bf  = (__bf16*)alloc((size_t)NQ * D * 2);
  __bf16* fi_bf  = (__bf16*)alloc((size_t)NI * D * 2);
  __bf16* ft_bf  = (__bf16*)alloc((size_t)NT * D * 2);
  __bf16* Wt_bf  = (__bf16*)alloc((size_t)D * D * 2);      // W^T, bf16
  __bf16* Qt_bf  = (__bf16*)alloc((size_t)OUT * OUT * 2);  // Q^T, bf16
  __bf16* Wq_bf  = (__bf16*)alloc((size_t)OUT * K2 * 2);
  __bf16* Wi_bf  = (__bf16*)alloc((size_t)OUT * K2 * 2);
  __bf16* Wt2_bf = (__bf16*)alloc((size_t)OUT * K2 * 2);
  float*  XW_q   = (float*)alloc((size_t)NQ * D * 4);      // feature_q @ W
  float*  XW_i   = (float*)alloc((size_t)NI * D * 4);      // feature_i @ W
  float*  XW_t   = (float*)alloc((size_t)NT * D * 4);      // feature_t @ W
  float*  acc_q  = (float*)alloc((size_t)NQ * D * 4);
  float*  acc_i  = (float*)alloc((size_t)NI * D * 4);
  float*  acc_t  = (float*)alloc((size_t)NT * D * 4);
  __bf16* cat_q  = (__bf16*)alloc((size_t)NQ * K2 * 2);
  __bf16* cat_i  = (__bf16*)alloc((size_t)NI * K2 * 2);
  __bf16* cat_t  = (__bf16*)alloc((size_t)NT * K2 * 2);
  float*  h_q    = (float*)alloc((size_t)NQ * OUT * 4);
  float*  h_i    = (float*)alloc((size_t)NI * OUT * 4);
  float*  h_t    = (float*)alloc((size_t)NT * OUT * 4);
  float*  stats  = (float*)alloc((size_t)2 * OUT * 4);
  __bf16* e_q    = (__bf16*)alloc((size_t)NQ * OUT * 2);
  __bf16* e_i    = (__bf16*)alloc((size_t)NI * OUT * 2);
  __bf16* e_t    = (__bf16*)alloc((size_t)NT * OUT * 2);
  __bf16* eQq    = (__bf16*)alloc((size_t)NQ * OUT * 2);
  __bf16* eQi    = (__bf16*)alloc((size_t)NI * OUT * 2);

  auto b1d = [](long long n) { return (unsigned)((n + 255) / 256); };
  dim3 blk(256);

  // 1) Precision conversion / weight transposes
  cvt_f2bf<<<b1d((long long)NQ * D), blk, 0, stream>>>(fq, fq_bf, NQ * D);
  cvt_f2bf<<<b1d((long long)NI * D), blk, 0, stream>>>(fi, fi_bf, NI * D);
  cvt_f2bf<<<b1d((long long)NT * D), blk, 0, stream>>>(ft, ft_bf, NT * D);
  transpose_f2bf<<<b1d((long long)D * D), blk, 0, stream>>>(W, Wt_bf, D, D);
  transpose_f2bf<<<b1d((long long)OUT * OUT), blk, 0, stream>>>(Q, Qt_bf, OUT, OUT);
  cvt_f2bf<<<b1d((long long)OUT * K2), blk, 0, stream>>>(Wq, Wq_bf, OUT * K2);
  cvt_f2bf<<<b1d((long long)OUT * K2), blk, 0, stream>>>(Wi, Wi_bf, OUT * K2);
  cvt_f2bf<<<b1d((long long)OUT * K2), blk, 0, stream>>>(Wt, Wt2_bf, OUT * K2);

  hipMemsetAsync(acc_q, 0, (size_t)NQ * D * 4, stream);
  hipMemsetAsync(acc_i, 0, (size_t)NI * D * 4, stream);
  hipMemsetAsync(acc_t, 0, (size_t)NT * D * 4, stream);

  // 2) feature @ W  (WMMA)
  gemm_nt_bf16<<<dim3(D / 64, NQ / 128), blk, 0, stream>>>(fq_bf, Wt_bf, XW_q, nullptr, nullptr, NQ, D, D);
  gemm_nt_bf16<<<dim3(D / 64, NI / 128), blk, 0, stream>>>(fi_bf, Wt_bf, XW_i, nullptr, nullptr, NI, D, D);
  gemm_nt_bf16<<<dim3(D / 64, NT / 128), blk, 0, stream>>>(ft_bf, Wt_bf, XW_t, nullptr, nullptr, NT, D, D);

  // 3) SPMM scatter-adds
  spmm_scatter<<<b1d((long long)EQI * 64), blk, 0, stream>>>(qi_src, qi_dst, qi_val, XW_i, acc_q, EQI);
  spmm_scatter<<<b1d((long long)EQI * 64), blk, 0, stream>>>(qi_dst, qi_src, qi_val, XW_q, acc_i, EQI);
  spmm_scatter<<<b1d((long long)EIT * 64), blk, 0, stream>>>(it_src, it_dst, it_val, XW_t, acc_i, EIT);
  spmm_scatter<<<b1d((long long)EIT * 64), blk, 0, stream>>>(it_dst, it_src, it_val, XW_i, acc_t, EIT);

  // 4) relu(hidden) ++ feature -> bf16 concat buffers
  pack_cat<<<b1d((long long)NQ * D), blk, 0, stream>>>(acc_q, fq, cat_q, NQ);
  pack_cat<<<b1d((long long)NI * D), blk, 0, stream>>>(acc_i, fi, cat_i, NI);
  pack_cat<<<b1d((long long)NT * D), blk, 0, stream>>>(acc_t, ft, cat_t, NT);

  // 5) h = cat @ Wl^T + b   (WMMA, bias epilogue)
  gemm_nt_bf16<<<dim3(OUT / 64, NQ / 128), blk, 0, stream>>>(cat_q, Wq_bf, h_q, nullptr, bq, NQ, OUT, K2);
  gemm_nt_bf16<<<dim3(OUT / 64, NI / 128), blk, 0, stream>>>(cat_i, Wi_bf, h_i, nullptr, bi, NI, OUT, K2);
  gemm_nt_bf16<<<dim3(OUT / 64, NT / 128), blk, 0, stream>>>(cat_t, Wt2_bf, h_t, nullptr, bt, NT, OUT, K2);

  // 6) BN (training-mode stats) + ReLU -> bf16 embeds (stats buffer reused serially)
  bn_stats<<<OUT, blk, 0, stream>>>(h_q, NQ, stats);
  bn_apply<<<b1d((long long)NQ * OUT), blk, 0, stream>>>(h_q, stats, gq, betaq, e_q, NQ);
  bn_stats<<<OUT, blk, 0, stream>>>(h_i, NI, stats);
  bn_apply<<<b1d((long long)NI * OUT), blk, 0, stream>>>(h_i, stats, gi, betai, e_i, NI);
  bn_stats<<<OUT, blk, 0, stream>>>(h_t, NT, stats);
  bn_apply<<<b1d((long long)NT * OUT), blk, 0, stream>>>(h_t, stats, gt, betat, e_t, NT);

  // 7) embed @ Q  (bf16 output for chaining)
  gemm_nt_bf16<<<dim3(OUT / 64, NQ / 128), blk, 0, stream>>>(e_q, Qt_bf, nullptr, eQq, nullptr, NQ, OUT, OUT);
  gemm_nt_bf16<<<dim3(OUT / 64, NI / 128), blk, 0, stream>>>(e_i, Qt_bf, nullptr, eQi, nullptr, NI, OUT, OUT);

  // 8) Scores: (eQ) @ embed^T  -> f32 outputs
  float* out = (float*)d_out;
  gemm_nt_bf16<<<dim3(NI / 64, NQ / 128), blk, 0, stream>>>(eQq, e_i, out, nullptr, nullptr, NQ, NI, OUT);
  gemm_nt_bf16<<<dim3(NT / 64, NI / 128), blk, 0, stream>>>(eQi, e_t, out + (size_t)NQ * NI, nullptr, nullptr, NI, NT, OUT);
}
